// CrossAttentionHead_81956565942657
// MI455X (gfx1250) — matrix operand order repaired and checked by reading, not compiled, not run on hardware
//
#include <hip/hip_runtime.h>
#include <hip/hip_bf16.h>
#include <math.h>

#define B_  8
#define S_  2048
#define D_  1024
#define HS_ 64

typedef __attribute__((ext_vector_type(16))) __bf16 v16bf;
typedef __attribute__((ext_vector_type(8)))  __bf16 v8bf;
typedef __attribute__((ext_vector_type(8)))  float  v8f;

// ---- A-operand packer: 16-bit A 16x32, contiguous k order (same as B operand):
// lane<16 holds K = 0..15, lane>=16 holds K = 16..31 (p1 = p0 + 8)
__device__ __forceinline__ v16bf pack_a(const __bf16* p0, const __bf16* p1) {
  v8bf lo = *(const v8bf*)p0;
  v8bf hi = *(const v8bf*)p1;
  v16bf r;
#pragma unroll
  for (int i = 0; i < 8; ++i) { r[i] = lo[i]; r[8 + i] = hi[i]; }
  return r;
}

__device__ __forceinline__ v8f wmma_bf16(v16bf a, v16bf b, v8f c) {
  return __builtin_amdgcn_wmma_f32_16x16x32_bf16(
      /*neg_a=*/false, a, /*neg_b=*/false, b,
      /*c_mod=*/(short)0, c, /*reuse_a=*/false, /*reuse_b=*/false);
}

// ---- gfx1250 async global->LDS copy (ASYNCcnt path, bypasses VGPRs)
__device__ __forceinline__ uint32_t lds_addr32(const void* p) {
  return (uint32_t)(size_t)(__attribute__((address_space(3))) const void*)p;
}
__device__ __forceinline__ void async_load_b128(uint32_t lds, const void* g) {
  asm volatile("global_load_async_to_lds_b128 %0, %1, off"
               :: "v"(lds), "v"((uint64_t)(size_t)g)
               : "memory");
}

// ============================================================================
// Projection: Out = X[16384 x 1024](f32) @ W[1024 x 64](f32), bf16 out.
// TRANSPOSED=false -> Out[b*S+s][h] row-major (q, k)
// TRANSPOSED=true  -> Out[b][h][s]           (vT, enables async B128 staging)
// Compile-time flag: keeps the unrolled store epilogue branch-free.
// ============================================================================
template <bool TRANSPOSED>
__global__ __launch_bounds__(256)
void proj_kernel(const float* __restrict__ X, const float* __restrict__ W,
                 __bf16* __restrict__ Out) {
  __shared__ __bf16 ldsWT[HS_ * 32];   // [n][kk] : W chunk transposed, bf16

  const int tid    = threadIdx.x;
  const int lane   = tid & 31;
  const int wave   = tid >> 5;
  const int lane16 = lane & 15;
  const int hi8    = (lane < 16) ? 0 : 8;    // accumulator row offset (M = j + hi8)
  const int kofs   = (lane < 16) ? 0 : 16;   // B-operand K base (sequential)
  const int rowbase = blockIdx.x * 128 + wave * 16;
  const int arow    = rowbase + lane16;      // A: lane holds row M=lane16

  v8f acc[4] = {};

  for (int kc = 0; kc < D_; kc += 32) {
    // Stage W chunk transposed to bf16: ldsWT[n*32+kk] = W[(kc+kk)*64+n]
    for (int i = tid; i < 32 * HS_; i += 256) {
      int n  = i & (HS_ - 1);
      int kk = i >> 6;
      ldsWT[n * 32 + kk] = (__bf16)W[(size_t)(kc + kk) * HS_ + n];
    }
    __syncthreads();

    // A operand from global fp32 (convert to bf16 in-flight)
    const float* xp = X + (size_t)arow * D_ + kc + kofs;
    float4 f0 = *(const float4*)(xp);
    float4 f1 = *(const float4*)(xp + 4);
    float4 f2 = *(const float4*)(xp + 8);
    float4 f3 = *(const float4*)(xp + 12);
    v16bf a;
    a[0]=(__bf16)f0.x; a[1]=(__bf16)f0.y; a[2]=(__bf16)f0.z; a[3]=(__bf16)f0.w;
    a[4]=(__bf16)f1.x; a[5]=(__bf16)f1.y; a[6]=(__bf16)f1.z; a[7]=(__bf16)f1.w;
    a[8]=(__bf16)f2.x; a[9]=(__bf16)f2.y; a[10]=(__bf16)f2.z; a[11]=(__bf16)f2.w;
    a[12]=(__bf16)f3.x; a[13]=(__bf16)f3.y; a[14]=(__bf16)f3.z; a[15]=(__bf16)f3.w;

#pragma unroll
    for (int t = 0; t < 4; ++t) {
      v16bf b = *(const v16bf*)(ldsWT + (16 * t + lane16) * 32 + kofs);
      acc[t] = wmma_bf16(a, b, acc[t]);
    }
    __syncthreads();
  }

  // C layout: VGPR j holds (M = j + hi8, N = lane16)
#pragma unroll
  for (int t = 0; t < 4; ++t)
#pragma unroll
    for (int j = 0; j < 8; ++j) {
      int grow = rowbase + j + hi8;
      int col  = 16 * t + lane16;
      __bf16 v = (__bf16)acc[t][j];
      if (TRANSPOSED)
        Out[((size_t)(grow >> 11) * HS_ + col) * S_ + (grow & (S_ - 1))] = v;
      else
        Out[(size_t)grow * HS_ + col] = v;
    }
}

// ============================================================================
// Flash attention fwd, causal. 128 threads = 4 waves; wave owns 16 q rows.
// Double-buffered async global->LDS staging of K tile + V^T tile.
// ============================================================================
__device__ __forceinline__ void stage_chunk_async(
    const __bf16* __restrict__ Kg, const __bf16* __restrict__ VTg,
    __bf16* ldsK, __bf16* ldsVT, int tid, int b, int kc) {
  // K tile [32 kk][64 h] bf16 = 256 x 16B segments; 2 per thread
#pragma unroll
  for (int it = 0; it < 2; ++it) {
    int seg = tid + it * 128;
    int kk = seg >> 3, part = seg & 7;
    async_load_b128(lds_addr32(ldsK + kk * HS_ + part * 8),
                    Kg + ((size_t)b * S_ + kc + kk) * HS_ + part * 8);
  }
  // V^T tile [64 h][32 kk] bf16 = 256 x 16B segments; 2 per thread
#pragma unroll
  for (int it = 0; it < 2; ++it) {
    int seg = tid + it * 128;
    int h = seg >> 2, part = seg & 3;
    async_load_b128(lds_addr32(ldsVT + h * 32 + part * 8),
                    VTg + ((size_t)b * HS_ + h) * S_ + kc + part * 8);
  }
}

__global__ __launch_bounds__(128)
void fattn_kernel(const __bf16* __restrict__ Q, const __bf16* __restrict__ K,
                  const __bf16* __restrict__ VT, float* __restrict__ Out) {
  __shared__ __bf16 ldsK[2][32 * HS_];    // [buf][kk][h]
  __shared__ __bf16 ldsVT[2][HS_ * 32];   // [buf][h][kk]
  __shared__ __bf16 ldsP[4][16 * 32];     // per-wave P staging (C->A transpose)

  const int tid    = threadIdx.x;
  const int lane   = tid & 31;
  const int wave   = tid >> 5;
  const int lane16 = lane & 15;
  const int hi8    = (lane < 16) ? 0 : 8;
  const int kofs   = (lane < 16) ? 0 : 16;

  const int b      = blockIdx.y;
  const int qblock = blockIdx.x * 64;
  const int qbase  = qblock + wave * 16;

  // Persistent Q A-operands (two K=32 slabs over HS=64)
  const __bf16* qrow = Q + ((size_t)b * S_ + qbase + lane16) * HS_;
  v16bf qa0 = pack_a(qrow + kofs,      qrow + kofs + 8);
  v16bf qa1 = pack_a(qrow + 32 + kofs, qrow + 32 + kofs + 8);

  v8f o[4] = {};
  float m[8], l[8];
#pragma unroll
  for (int j = 0; j < 8; ++j) { m[j] = -INFINITY; l[j] = 0.0f; }

  const float kscale = 0.125f * 1.44269504088896f;  // 1/sqrt(64) * log2(e)
  const int nch = (qblock + 64) / 32;               // causal chunk count

  // prologue: prefetch chunk 0 into buffer 0
  stage_chunk_async(K, VT, ldsK[0], ldsVT[0], tid, b, 0);

  for (int c = 0; c < nch; ++c) {
    const int cur = c & 1;
    const int kc  = c * 32;
    const bool more = (c + 1 < nch);
    if (more)  // prefetch next chunk into the other buffer (freed last iter)
      stage_chunk_async(K, VT, ldsK[cur ^ 1], ldsVT[cur ^ 1], tid, b, kc + 32);
    // async loads retire in order: <=4 outstanding leaves only next chunk's
    if (more) asm volatile("s_wait_asynccnt 0x4" ::: "memory");
    else      asm volatile("s_wait_asynccnt 0x0" ::: "memory");
    __syncthreads();   // all waves' cur-chunk data visible in LDS

    if (kc <= qbase + 15) {   // chunk intersects this wave's causal window
      // ---- S = Q K^T (scores for 32 keys, two 16-col tiles)
      v8f s[2] = {};
#pragma unroll
      for (int nh = 0; nh < 2; ++nh) {
        const __bf16* kp = ldsK[cur] + (16 * nh + lane16) * HS_;
        v16bf b0 = *(const v16bf*)(kp + kofs);
        v16bf b1 = *(const v16bf*)(kp + 32 + kofs);
        s[nh] = wmma_bf16(qa0, b0, s[nh]);
        s[nh] = wmma_bf16(qa1, b1, s[nh]);
      }
      // ---- scale + causal mask (C layout: row=j+hi8, col=lane16)
#pragma unroll
      for (int nh = 0; nh < 2; ++nh) {
        int keyn = kc + 16 * nh + lane16;
#pragma unroll
        for (int j = 0; j < 8; ++j) {
          int qn = qbase + j + hi8;
          float vsc = s[nh][j] * kscale;
          s[nh][j] = (keyn > qn) ? -INFINITY : vsc;
        }
      }
      // ---- online softmax (row reductions across 16-lane halves)
      float pa[2][8];
#pragma unroll
      for (int j = 0; j < 8; ++j) {
        float mx = fmaxf(s[0][j], s[1][j]);
#pragma unroll
        for (int off = 1; off < 16; off <<= 1)
          mx = fmaxf(mx, __shfl_xor(mx, off, 32));
        float mn   = fmaxf(m[j], mx);
        float corr = __builtin_exp2f(m[j] - mn);   // 0 on first chunk
        m[j] = mn;
        float p0 = __builtin_exp2f(s[0][j] - mn);
        float p1 = __builtin_exp2f(s[1][j] - mn);
        pa[0][j] = p0; pa[1][j] = p1;
        float rs = p0 + p1;
#pragma unroll
        for (int off = 1; off < 16; off <<= 1)
          rs += __shfl_xor(rs, off, 32);
        l[j] = l[j] * corr + rs;
#pragma unroll
        for (int t = 0; t < 4; ++t) o[t][j] *= corr;
      }
      // ---- C-layout -> A-layout transpose of P via per-wave LDS
      __bf16* pw = ldsP[wave];
#pragma unroll
      for (int nh = 0; nh < 2; ++nh)
#pragma unroll
        for (int j = 0; j < 8; ++j)
          pw[(j + hi8) * 32 + 16 * nh + lane16] = (__bf16)pa[nh][j];
      asm volatile("s_wait_dscnt 0x0" ::: "memory");  // wave-local store->load
      const __bf16* pl = pw + lane16 * 32 + kofs;
      v16bf aP = pack_a(pl, pl + 8);
      // ---- O += P V  (4 column tiles over HS=64)
#pragma unroll
      for (int t = 0; t < 4; ++t) {
        v16bf bV = *(const v16bf*)(ldsVT[cur] + (16 * t + lane16) * 32 + kofs);
        o[t] = wmma_bf16(aP, bV, o[t]);
      }
    }
    __syncthreads();   // cur buffer free for prefetch next iteration
  }

  // ---- normalize and write fp32 output
#pragma unroll
  for (int j = 0; j < 8; ++j) {
    float rl = 1.0f / l[j];
#pragma unroll
    for (int t = 0; t < 4; ++t)
      Out[((size_t)b * S_ + qbase + j + hi8) * HS_ + 16 * t + lane16] =
          o[t][j] * rl;
  }
}

extern "C" void kernel_launch(void* const* d_in, const int* in_sizes, int n_in,
                              void* d_out, int out_size, void* d_ws, size_t ws_size,
                              hipStream_t stream) {
  const float* query = (const float*)d_in[0];
  const float* key   = (const float*)d_in[1];
  const float* value = (const float*)d_in[2];
  const float* Wq    = (const float*)d_in[3];
  const float* Wk    = (const float*)d_in[4];
  const float* Wv    = (const float*)d_in[5];
  float* out = (float*)d_out;

  // bf16 q/k/vT workspace: 3 * 8*2048*64 * 2B = 6 MB
  __bf16* qws  = (__bf16*)d_ws;
  __bf16* kws  = qws + (size_t)B_ * S_ * HS_;
  __bf16* vtws = kws + (size_t)B_ * S_ * HS_;

  dim3 pgrid(B_ * S_ / 128, 1, 1);
  proj_kernel<false><<<pgrid, 256, 0, stream>>>(query, Wq, qws);
  proj_kernel<false><<<pgrid, 256, 0, stream>>>(key,   Wk, kws);
  proj_kernel<true ><<<pgrid, 256, 0, stream>>>(value, Wv, vtws);  // V^T

  dim3 agrid(S_ / 64, B_, 1);
  fattn_kernel<<<agrid, 128, 0, stream>>>(qws, kws, vtws, out);
}
